// BertEncoder_27590869909847
// MI455X (gfx1250) — matrix operand order repaired and checked by reading, not compiled, not run on hardware
//
#include <hip/hip_runtime.h>

// ---------------------------------------------------------------------------
// Problem constants (match reference)
// ---------------------------------------------------------------------------
#define H_DIM   768
#define NH      12
#define DH      64
#define FF_DIM  3072
#define BATCH   16
#define SEQ     512
#define TOK     (BATCH * SEQ)        // 8192
#define PARTS   6
#define LN_EPS  1e-12f

// ---------------------------------------------------------------------------
// CDNA5 WMMA / TDM types
// ---------------------------------------------------------------------------
typedef __attribute__((ext_vector_type(16))) __bf16 bf16x16;
typedef __attribute__((ext_vector_type(8)))  __bf16 bf16x8;
typedef __attribute__((ext_vector_type(8)))  float  f32x8;
typedef __attribute__((ext_vector_type(4)))  unsigned int v4u;
typedef __attribute__((ext_vector_type(8)))  int v8i;
typedef __attribute__((ext_vector_type(4)))  int v4i;

__device__ __forceinline__ f32x8 zero8() {
    f32x8 z;
#pragma unroll
    for (int i = 0; i < 8; ++i) z[i] = 0.0f;
    return z;
}

__device__ __forceinline__ bf16x8 ld8(const __bf16* p) {
    return *(const bf16x8*)p;
}

__device__ __forceinline__ bf16x16 mk16(bf16x8 lo, bf16x8 hi) {
    bf16x16 r;
#pragma unroll
    for (int i = 0; i < 8; ++i) { r[i] = lo[i]; r[8 + i] = hi[i]; }
    return r;
}

__device__ __forceinline__ f32x8 wmma_bf16(bf16x16 a, bf16x16 b, f32x8 c) {
    // D = A(16x32 bf16) * B(32x16 bf16) + C(16x16 f32)
    return __builtin_amdgcn_wmma_f32_16x16x32_bf16(
        /*neg_a=*/false, a, /*neg_b=*/false, b,
        /*c_mod=*/(short)0, c, /*reuse_a=*/false, /*reuse_b=*/false);
}

__device__ __forceinline__ float gelu_exact(float x) {
    return 0.5f * x * (1.0f + erff(x * 0.70710678118654752f));
}

// ---------------------------------------------------------------------------
// Tensor Data Mover: DMA a 2D bf16 tile [rows x 32] (row stride = K elements)
// from global into LDS. Issued once per wave; tracked by TENSORcnt.
// Toolchain here is the 6-arg builtin (clang-23 / therock-10.0 headers):
//   (uint32x4 g0, int32x8 g1, int32x4 g2, int32x4 g3, int32x8 g4, i32 cpol)
// ---------------------------------------------------------------------------
#if __has_builtin(__builtin_amdgcn_tensor_load_to_lds)
#define HAVE_TDM 1
__device__ __forceinline__ void tdm_load_tile_2d(unsigned lds_off,
                                                 const void* gaddr,
                                                 unsigned tile_w_elts,
                                                 unsigned tile_h_rows,
                                                 unsigned tensor_w_elts,
                                                 unsigned tensor_h_rows) {
    unsigned long long ga = (unsigned long long)(size_t)gaddr;   // byte address
    v4u g0;
    g0[0] = 1u;                                   // count=1, user desc
    g0[1] = lds_off;                              // lds_addr (bytes)
    g0[2] = (unsigned)(ga & 0xFFFFFFFFull);       // global_addr[31:0]
    g0[3] = (unsigned)((ga >> 32) & 0x1FFFFFFull) // global_addr[56:32]
            | (2u << 30);                         // type = 2 ("image")
    v8i g1;
    g1[0] = (int)(1u << 16);                      // wg_mask=0, data_size=1 (2B)
    g1[1] = (int)((tensor_w_elts & 0xFFFFu) << 16);        // tensor_dim0 lo16
    g1[2] = (int)((tensor_w_elts >> 16) |
                  ((tensor_h_rows & 0xFFFFu) << 16));      // dim0 hi | dim1 lo
    g1[3] = (int)((tensor_h_rows >> 16) |
                  ((tile_w_elts & 0xFFFFu) << 16));        // dim1 hi | tile_dim0
    g1[4] = (int)(tile_h_rows & 0xFFFFu);                  // tile_dim1 (tile_dim2=0)
    g1[5] = (int)tensor_w_elts;                            // tensor_dim0_stride[31:0]
    g1[6] = 0;                                             // stride0 hi | stride1 lo
    g1[7] = 0;
    v4i gz;
    gz[0] = 0; gz[1] = 0; gz[2] = 0; gz[3] = 0;
    v8i gz8;
#pragma unroll
    for (int i = 0; i < 8; ++i) gz8[i] = 0;
    __builtin_amdgcn_tensor_load_to_lds(g0, g1, gz, gz, gz8, 0);
}
#else
#define HAVE_TDM 0
#endif

// ---------------------------------------------------------------------------
// fp32 -> bf16 streaming convert
// ---------------------------------------------------------------------------
__global__ void cvt_f32_bf16(const float* __restrict__ src,
                             __bf16* __restrict__ dst, long n) {
    long i = (long)blockIdx.x * blockDim.x + threadIdx.x;
    long stride = (long)gridDim.x * blockDim.x;
    for (; i < n; i += stride) dst[i] = (__bf16)src[i];
}

// ---------------------------------------------------------------------------
// GEMM: C[M,N] = act(A[M,K]bf16 @ B[K,N]bf16 + bias[N]); fp32 and/or bf16 out.
// M,N,K multiples of 128/32. Block 128x128, BK=32, 256 threads = 8 waves.
// A tile staged by the Tensor Data Mover (wave 0 issues, TENSORcnt-synced);
// B tile transposed into LDS so fragments are aligned ds_load_b128 pairs.
// ---------------------------------------------------------------------------
__global__ __launch_bounds__(256)
void gemm_bf16(const __bf16* __restrict__ A, const __bf16* __restrict__ Bw,
               const float* __restrict__ bias,
               float* __restrict__ C, __bf16* __restrict__ Cbf,
               int M, int N, int K, int act) {
    __shared__ __attribute__((aligned(16))) __bf16 As[128 * 32];   // [m][k]
    __shared__ __attribute__((aligned(16))) __bf16 Bs[128 * 32];   // [n][k] (transposed)

    const int tid  = threadIdx.x;
    const int lane = tid & 31;
    const int w    = tid >> 5;
    const int wm   = w & 1;        // 2 wave rows  (64 M each)
    const int wn   = w >> 1;       // 4 wave cols  (32 N each)
    const int half = lane >> 4;    // 0: lanes 0-15, 1: lanes 16-31
    const int l16  = lane & 15;

    const int m0 = blockIdx.y * 128;
    const int n0 = blockIdx.x * 128;

#if HAVE_TDM
    const unsigned as_off = (unsigned)(size_t)(&As[0]);   // LDS byte offset
#endif

    f32x8 acc[4][2];
#pragma unroll
    for (int i = 0; i < 4; ++i)
#pragma unroll
        for (int j = 0; j < 2; ++j) acc[i][j] = zero8();

    for (int k0 = 0; k0 < K; k0 += 32) {
        // ---- stage A tile (128x32) ----
#if HAVE_TDM
        if (w == 0) {
            tdm_load_tile_2d(as_off, A + (size_t)m0 * K + k0,
                             /*tile_w=*/32, /*tile_h=*/128,
                             /*tensor_w=*/(unsigned)K, /*tensor_h=*/(unsigned)M);
        }
#else
#pragma unroll
        for (int it = 0; it < 2; ++it) {
            int c  = tid + it * 256;          // 0..511
            int m  = c >> 2;                  // 0..127
            int k8 = (c & 3) * 8;             // 0,8,16,24
            const __bf16* gp = A + (size_t)(m0 + m) * K + k0 + k8;
            *(uint4*)(&As[m * 32 + k8]) = *(const uint4*)gp;
        }
#endif
        // ---- stage B tile transposed (Bs[n][k]) ----
#pragma unroll
        for (int it = 0; it < 2; ++it) {
            int c  = tid + it * 256;          // 0..511
            int kk = c >> 4;                  // 0..31
            int n8 = (c & 15) * 8;            // 0..120
            const __bf16* gp = Bw + (size_t)(k0 + kk) * N + n0 + n8;
            __bf16 tmp[8];
            *(uint4*)tmp = *(const uint4*)gp;
            if (k0 + 32 < K) __builtin_prefetch(gp + (size_t)32 * N, 0, 1);
#pragma unroll
            for (int j = 0; j < 8; ++j) Bs[(n8 + j) * 32 + kk] = tmp[j];
        }
#if HAVE_TDM
        if (w == 0) __builtin_amdgcn_s_wait_tensorcnt(0);
#endif
        __syncthreads();

        // ---- fragments ----
        const int akb = half ? 16 : 0;        // A: 16 contiguous K per half-wave
        bf16x16 afr[4];
#pragma unroll
        for (int mt = 0; mt < 4; ++mt) {
            int row = wm * 64 + mt * 16 + l16;
            afr[mt] = mk16(ld8(&As[row * 32 + akb]),
                           ld8(&As[row * 32 + akb + 8]));
        }
        const int bkb = half ? 16 : 0;        // B: 16 contiguous K per half-wave
        bf16x16 bfr[2];
#pragma unroll
        for (int nt = 0; nt < 2; ++nt) {
            int col = wn * 32 + nt * 16 + l16;
            bfr[nt] = mk16(ld8(&Bs[col * 32 + bkb]),
                           ld8(&Bs[col * 32 + bkb + 8]));
        }

#pragma unroll
        for (int mt = 0; mt < 4; ++mt)
#pragma unroll
            for (int nt = 0; nt < 2; ++nt)
                acc[mt][nt] = wmma_bf16(afr[mt], bfr[nt], acc[mt][nt]);

        __syncthreads();
    }

    // ---- epilogue: D layout -> global ----
#pragma unroll
    for (int mt = 0; mt < 4; ++mt) {
#pragma unroll
        for (int nt = 0; nt < 2; ++nt) {
            int col = n0 + wn * 32 + nt * 16 + l16;
            float bv = bias ? bias[col] : 0.0f;
#pragma unroll
            for (int r = 0; r < 8; ++r) {
                int row = m0 + wm * 64 + mt * 16 + r + 8 * half;
                float val = acc[mt][nt][r] + bv;
                if (act == 1) val = gelu_exact(val);
                size_t idx = (size_t)row * N + col;
                if (C)   C[idx]   = val;
                if (Cbf) Cbf[idx] = (__bf16)val;
            }
        }
    }
}

// ---------------------------------------------------------------------------
// Flash attention: block = (b, head, 64 q rows), 128 threads = 4 waves.
// q,k,v: bf16 [TOK, H] (head-major inside H). ctx out: bf16 [TOK, H].
// ---------------------------------------------------------------------------
__global__ __launch_bounds__(128)
void flash_attn(const __bf16* __restrict__ q, const __bf16* __restrict__ k,
                const __bf16* __restrict__ v, const float* __restrict__ mask,
                __bf16* __restrict__ ctx) {
    __shared__ __attribute__((aligned(16))) __bf16 Ks[32 * 64];      // [key][d]
    __shared__ __attribute__((aligned(16))) __bf16 Vt[64 * 32];      // [d][key]
    __shared__ __attribute__((aligned(16))) __bf16 Ps[4][16 * 32];   // per-wave P

    const int tid  = threadIdx.x;
    const int lane = tid & 31;
    const int w    = tid >> 5;
    const int half = lane >> 4;
    const int l16  = lane & 15;

    const int b    = blockIdx.z;
    const int head = blockIdx.y;
    const int q0   = blockIdx.x * 64 + w * 16;     // 16 q rows per wave
    const size_t hoff = (size_t)head * DH;

    // Q fragments (16 x 64 -> two 16x32 A-frags), loaded straight from global
    const int akb = half ? 16 : 0;
    bf16x16 qf[2];
#pragma unroll
    for (int f = 0; f < 2; ++f) {
        const __bf16* qr = q + ((size_t)(b * SEQ + q0 + l16) * H_DIM) + hoff + f * 32;
        qf[f] = mk16(ld8(qr + akb), ld8(qr + akb + 8));
    }

    f32x8 acc[4];
#pragma unroll
    for (int nt = 0; nt < 4; ++nt) acc[nt] = zero8();
    float mrow[8], srow[8];
#pragma unroll
    for (int r = 0; r < 8; ++r) { mrow[r] = -1e30f; srow[r] = 0.0f; }

    for (int kb = 0; kb < SEQ / 32; ++kb) {
        __syncthreads();
        // ---- stage K (natural) and V (transposed) tiles: 32 keys x 64 d ----
        {
            int key = tid >> 2;                 // 0..31
            int d16 = (tid & 3) * 16;           // 0,16,32,48
            const __bf16* kg = k + ((size_t)(b * SEQ + kb * 32 + key) * H_DIM) + hoff + d16;
            const __bf16* vg = v + ((size_t)(b * SEQ + kb * 32 + key) * H_DIM) + hoff + d16;
            *(uint4*)(&Ks[key * 64 + d16])     = *(const uint4*)kg;
            *(uint4*)(&Ks[key * 64 + d16 + 8]) = *(const uint4*)(kg + 8);
            __bf16 tv[16];
            *(uint4*)tv       = *(const uint4*)vg;
            *(uint4*)(tv + 8) = *(const uint4*)(vg + 8);
#pragma unroll
            for (int j = 0; j < 16; ++j) Vt[(d16 + j) * 32 + key] = tv[j];
        }
        __syncthreads();

        // ---- scores: two 16(q) x 16(key) tiles, contraction over DH=64 ----
        const int dbase = half ? 16 : 0;
        f32x8 st[2];
#pragma unroll
        for (int kt = 0; kt < 2; ++kt) {
            int key = kt * 16 + l16;
            st[kt] = zero8();
#pragma unroll
            for (int dc = 0; dc < 2; ++dc) {
                bf16x16 bfr = mk16(ld8(&Ks[key * 64 + dc * 32 + dbase]),
                                   ld8(&Ks[key * 64 + dc * 32 + dbase + 8]));
                st[kt] = wmma_bf16(qf[dc], bfr, st[kt]);
            }
        }
        float mv0 = mask[(size_t)b * SEQ + kb * 32 + l16];
        float mv1 = mask[(size_t)b * SEQ + kb * 32 + 16 + l16];

        // ---- online softmax + P staging (per-wave LDS patch) ----
        __bf16* Pw = Ps[w];
#pragma unroll
        for (int r = 0; r < 8; ++r) {
            float c0 = st[0][r] * 0.125f + mv0;    // 1/sqrt(64)
            float c1 = st[1][r] * 0.125f + mv1;
            float mx = fmaxf(c0, c1);
#pragma unroll
            for (int o = 1; o < 16; o <<= 1) mx = fmaxf(mx, __shfl_xor(mx, o, 32));
            float nm = fmaxf(mrow[r], mx);
            float al = __expf(mrow[r] - nm);
            float p0 = __expf(c0 - nm);
            float p1 = __expf(c1 - nm);
            float rs = p0 + p1;
#pragma unroll
            for (int o = 1; o < 16; o <<= 1) rs += __shfl_xor(rs, o, 32);
            srow[r] = srow[r] * al + rs;
            mrow[r] = nm;
#pragma unroll
            for (int nt = 0; nt < 4; ++nt) acc[nt][r] *= al;
            int prow = r + 8 * half;
            Pw[prow * 32 + l16]      = (__bf16)p0;
            Pw[prow * 32 + 16 + l16] = (__bf16)p1;
        }
        // same-wave DS ops are in-order (ISA 7.3); compiler inserts dscnt waits
        bf16x16 pf = mk16(ld8(&Pw[l16 * 32 + akb]),
                          ld8(&Pw[l16 * 32 + akb + 8]));

        // ---- ctx += P(16x32) x V(32x64) ----
        const int keyb = half ? 16 : 0;
#pragma unroll
        for (int nt = 0; nt < 4; ++nt) {
            int d = nt * 16 + l16;
            bf16x16 vfr = mk16(ld8(&Vt[d * 32 + keyb]),
                               ld8(&Vt[d * 32 + keyb + 8]));
            acc[nt] = wmma_bf16(pf, vfr, acc[nt]);
        }
    }

    // ---- normalize + store ctx (bf16) ----
#pragma unroll
    for (int nt = 0; nt < 4; ++nt) {
#pragma unroll
        for (int r = 0; r < 8; ++r) {
            int row = q0 + r + 8 * half;
            float val = acc[nt][r] / srow[r];
            ctx[((size_t)(b * SEQ + row) * H_DIM) + hoff + nt * 16 + l16] = (__bf16)val;
        }
    }
}

// ---------------------------------------------------------------------------
// Fused residual-add + LayerNorm; one block per token, 256 threads over 768.
// ---------------------------------------------------------------------------
__global__ __launch_bounds__(256)
void ln_kernel(const float* __restrict__ x, const float* __restrict__ res,
               const float* __restrict__ sc, const float* __restrict__ bi,
               float* __restrict__ y, __bf16* __restrict__ ybf) {
    __shared__ float sred[8];
    const int t = blockIdx.x, tid = threadIdx.x;
    const int lane = tid & 31, w = tid >> 5;
    const size_t base = (size_t)t * H_DIM;

    float v[3];
#pragma unroll
    for (int j = 0; j < 3; ++j) {
        int i = tid + j * 256;
        v[j] = x[base + i] + res[base + i];
    }
    float s = v[0] + v[1] + v[2];
#pragma unroll
    for (int o = 1; o < 32; o <<= 1) s += __shfl_xor(s, o, 32);
    if (lane == 0) sred[w] = s;
    __syncthreads();
    float tot = 0.0f;
#pragma unroll
    for (int i = 0; i < 8; ++i) tot += sred[i];
    const float mean = tot * (1.0f / H_DIM);
    __syncthreads();

    float sq = 0.0f;
#pragma unroll
    for (int j = 0; j < 3; ++j) { float d = v[j] - mean; sq += d * d; }
#pragma unroll
    for (int o = 1; o < 32; o <<= 1) sq += __shfl_xor(sq, o, 32);
    if (lane == 0) sred[w] = sq;
    __syncthreads();
    float vtot = 0.0f;
#pragma unroll
    for (int i = 0; i < 8; ++i) vtot += sred[i];
    const float rs = rsqrtf(vtot * (1.0f / H_DIM) + LN_EPS);

#pragma unroll
    for (int j = 0; j < 3; ++j) {
        int i = tid + j * 256;
        float o = (v[j] - mean) * rs * sc[i] + bi[i];
        y[base + i] = o;
        if (ybf) ybf[base + i] = (__bf16)o;
    }
}

// ---------------------------------------------------------------------------
// Agent/pooler: pooled = tanh(h[:,0] @ p_w + p_b); logits; softmax; sample.
// One block per example.
// ---------------------------------------------------------------------------
__global__ __launch_bounds__(256)
void agent_kernel(const float* __restrict__ h, const float* __restrict__ pw,
                  const float* __restrict__ pb, const float* __restrict__ aw,
                  const float* __restrict__ ab,
                  float* __restrict__ probs_out, float* __restrict__ act_out_f,
                  int* __restrict__ act_ws, int part) {
    __shared__ float pooled[H_DIM];
    __shared__ float red[16];
    const int b = blockIdx.x, tid = threadIdx.x;
    const int lane = tid & 31, w = tid >> 5;
    const float* hrow = h + (size_t)b * SEQ * H_DIM;   // token 0

    for (int j = tid; j < H_DIM; j += 256) {
        float acc = pb[j];
        for (int kk = 0; kk < H_DIM; ++kk) acc += hrow[kk] * pw[(size_t)kk * H_DIM + j];
        pooled[j] = tanhf(acc);
    }
    __syncthreads();

    float l0 = 0.0f, l1 = 0.0f;
    for (int kk = tid; kk < H_DIM; kk += 256) {
        float p = pooled[kk];
        l0 += p * aw[kk * 2 + 0];
        l1 += p * aw[kk * 2 + 1];
    }
#pragma unroll
    for (int o = 1; o < 32; o <<= 1) { l0 += __shfl_xor(l0, o, 32); l1 += __shfl_xor(l1, o, 32); }
    if (lane == 0) { red[w * 2] = l0; red[w * 2 + 1] = l1; }
    __syncthreads();
    if (tid == 0) {
        float L0 = ab[0], L1 = ab[1];
#pragma unroll
        for (int i = 0; i < 8; ++i) { L0 += red[i * 2]; L1 += red[i * 2 + 1]; }
        float m = fmaxf(L0, L1);
        float e0 = __expf(L0 - m), e1 = __expf(L1 - m);
        float inv = 1.0f / (e0 + e1);
        float p0 = e0 * inv, p1 = e1 * inv;
        probs_out[b * 2 + 0] = p0;
        probs_out[b * 2 + 1] = p1;
        unsigned x = (unsigned)(part * 747796405u + b * 2891336453u + 12345u);
        x ^= x >> 17; x *= 0xED5AD4BBu; x ^= x >> 11; x *= 0xAC4C1B51u; x ^= x >> 15;
        float u = (float)(x >> 8) * (1.0f / 16777216.0f);
        int act = (u < p1) ? 1 : 0;
        act_ws[b] = act;
        act_out_f[b] = (float)act;
    }
}

// ---------------------------------------------------------------------------
// Per-example route select: h = act[b] ? large_h : base_h
// ---------------------------------------------------------------------------
__global__ void select_kernel(const int* __restrict__ act,
                              const float* __restrict__ lh,
                              const float* __restrict__ bh,
                              float* __restrict__ out, long n) {
    long i = (long)blockIdx.x * blockDim.x + threadIdx.x;
    long stride = (long)gridDim.x * blockDim.x;
    for (; i < n; i += stride) {
        int b = (int)(i / ((long)SEQ * H_DIM));
        out[i] = act[b] ? lh[i] : bh[i];
    }
}

// ---------------------------------------------------------------------------
// Host side
// ---------------------------------------------------------------------------
struct LayerW {
    const float *qw, *qb, *kw, *kb, *vw, *vb, *aow, *aob;
    const float *ln1s, *ln1b, *iw, *ib, *ow, *ob, *ln2s, *ln2b;
};

static LayerW get_layer(void* const* din, int base, int L) {
    LayerW p;
    const size_t HH = (size_t)H_DIM * H_DIM, HF = (size_t)H_DIM * FF_DIM;
    p.qw   = (const float*)din[base + 0]  + (size_t)L * HH;
    p.qb   = (const float*)din[base + 1]  + (size_t)L * H_DIM;
    p.kw   = (const float*)din[base + 2]  + (size_t)L * HH;
    p.kb   = (const float*)din[base + 3]  + (size_t)L * H_DIM;
    p.vw   = (const float*)din[base + 4]  + (size_t)L * HH;
    p.vb   = (const float*)din[base + 5]  + (size_t)L * H_DIM;
    p.aow  = (const float*)din[base + 6]  + (size_t)L * HH;
    p.aob  = (const float*)din[base + 7]  + (size_t)L * H_DIM;
    p.ln1s = (const float*)din[base + 8]  + (size_t)L * H_DIM;
    p.ln1b = (const float*)din[base + 9]  + (size_t)L * H_DIM;
    p.iw   = (const float*)din[base + 10] + (size_t)L * HF;
    p.ib   = (const float*)din[base + 11] + (size_t)L * FF_DIM;
    p.ow   = (const float*)din[base + 12] + (size_t)L * HF;
    p.ob   = (const float*)din[base + 13] + (size_t)L * H_DIM;
    p.ln2s = (const float*)din[base + 14] + (size_t)L * H_DIM;
    p.ln2b = (const float*)din[base + 15] + (size_t)L * H_DIM;
    return p;
}

extern "C" void kernel_launch(void* const* d_in, const int* in_sizes, int n_in,
                              void* d_out, int out_size, void* d_ws, size_t ws_size,
                              hipStream_t stream) {
    (void)in_sizes; (void)n_in; (void)out_size;

    const float* hidden = (const float*)d_in[0];
    const float* mask   = (const float*)d_in[1];
    const float* ag_pw  = (const float*)d_in[34];
    const float* ag_pb  = (const float*)d_in[35];
    const float* ag_aw  = (const float*)d_in[36];
    const float* ag_ab  = (const float*)d_in[37];
    float* out = (float*)d_out;

    // ---- workspace carve-up ----
    char* ws = (char*)d_ws;
    size_t cur = 0;
    auto alloc = [&](size_t bytes) -> char* {
        char* p = ws + cur;
        cur += (bytes + 255) & ~(size_t)255;
        return p;
    };
    const size_t TH_F  = (size_t)TOK * H_DIM * sizeof(float);
    const size_t TH_B  = (size_t)TOK * H_DIM * sizeof(__bf16);
    const size_t TF_B  = (size_t)TOK * FF_DIM * sizeof(__bf16);
    const size_t WB    = (size_t)H_DIM * FF_DIM * sizeof(__bf16);

    float*  h_cur   = (float*)alloc(TH_F);
    float*  base_h  = (float*)alloc(TH_F);
    float*  large_h = (float*)alloc(TH_F);
    float*  h_tmp   = (float*)alloc(TH_F);
    float*  af      = (float*)alloc(TH_F);
    float*  attnf   = (float*)alloc(TH_F);
    float*  of32    = (float*)alloc(TH_F);
    __bf16* hbf     = (__bf16*)alloc(TH_B);
    __bf16* qbf     = (__bf16*)alloc(TH_B);
    __bf16* kbf     = (__bf16*)alloc(TH_B);
    __bf16* vbf     = (__bf16*)alloc(TH_B);
    __bf16* ctxbf   = (__bf16*)alloc(TH_B);
    __bf16* attnbf  = (__bf16*)alloc(TH_B);
    __bf16* interbf = (__bf16*)alloc(TF_B);
    __bf16* wbuf    = (__bf16*)alloc(WB);
    int*    act_ws  = (int*)alloc(64 * sizeof(int));
    (void)ws_size;

    auto cvt = [&](const float* s, __bf16* d, long n) {
        int blocks = (int)((n + 255) / 256);
        if (blocks > 4096) blocks = 4096;
        cvt_f32_bf16<<<blocks, 256, 0, stream>>>(s, d, n);
    };
    auto gemm = [&](const __bf16* A, const __bf16* Bw, const float* bias,
                    float* C, __bf16* Cbf, int M, int N, int K, int act) {
        gemm_bf16<<<dim3(N / 128, M / 128), 256, 0, stream>>>(A, Bw, bias, C, Cbf, M, N, K, act);
    };

    auto run_layer = [&](const LayerW& p, const float* hin, float* hout) {
        cvt(hin, hbf, (long)TOK * H_DIM);
        // QKV projections (bf16 outputs only)
        cvt(p.qw, wbuf, (long)H_DIM * H_DIM);
        gemm(hbf, wbuf, p.qb, nullptr, qbf, TOK, H_DIM, H_DIM, 0);
        cvt(p.kw, wbuf, (long)H_DIM * H_DIM);
        gemm(hbf, wbuf, p.kb, nullptr, kbf, TOK, H_DIM, H_DIM, 0);
        cvt(p.vw, wbuf, (long)H_DIM * H_DIM);
        gemm(hbf, wbuf, p.vb, nullptr, vbf, TOK, H_DIM, H_DIM, 0);
        // attention
        flash_attn<<<dim3(SEQ / 64, NH, BATCH), 128, 0, stream>>>(qbf, kbf, vbf, mask, ctxbf);
        // attention output projection + LN1
        cvt(p.aow, wbuf, (long)H_DIM * H_DIM);
        gemm(ctxbf, wbuf, p.aob, af, nullptr, TOK, H_DIM, H_DIM, 0);
        ln_kernel<<<TOK, 256, 0, stream>>>(af, hin, p.ln1s, p.ln1b, attnf, attnbf);
        // FFN
        cvt(p.iw, wbuf, (long)H_DIM * FF_DIM);
        gemm(attnbf, wbuf, p.ib, nullptr, interbf, TOK, FF_DIM, H_DIM, 1 /*gelu*/);
        cvt(p.ow, wbuf, (long)FF_DIM * H_DIM);
        gemm(interbf, wbuf, p.ob, of32, nullptr, TOK, H_DIM, FF_DIM, 0);
        ln_kernel<<<TOK, 256, 0, stream>>>(of32, attnf, p.ln2s, p.ln2b, hout, nullptr);
    };

    // h_cur = hidden_states
    (void)hipMemcpyAsync(h_cur, hidden, TH_F, hipMemcpyDeviceToDevice, stream);

    float* probs_base = out + (size_t)TOK * H_DIM;           // [6,16,2]
    float* acts_base  = probs_base + PARTS * BATCH * 2;      // [6,16]

    for (int part = 0; part < PARTS; ++part) {
        agent_kernel<<<BATCH, 256, 0, stream>>>(
            h_cur, ag_pw, ag_pb, ag_aw, ag_ab,
            probs_base + part * BATCH * 2, acts_base + part * BATCH, act_ws, part);

        // base stack: 1 scc layer
        LayerW pb_ = get_layer(d_in, 18, part);
        run_layer(pb_, h_cur, base_h);

        // large stack: 2 prd layers
        LayerW pl0 = get_layer(d_in, 2, part * 2);
        LayerW pl1 = get_layer(d_in, 2, part * 2 + 1);
        run_layer(pl0, h_cur, h_tmp);
        run_layer(pl1, h_tmp, large_h);

        select_kernel<<<4096, 256, 0, stream>>>(act_ws, large_h, base_h, h_cur,
                                                (long)TOK * H_DIM);
    }

    (void)hipMemcpyAsync(out, h_cur, TH_F, hipMemcpyDeviceToDevice, stream);
}
